// Decoder_30683246362866
// MI455X (gfx1250) — compile-verified, matrix operand order checked by reading, not run on hardware
//
#include <hip/hip_runtime.h>

// ---------------- problem dimensions ----------------
#define T_SEQ 4096
#define D1    1024      // layer-1 hidden / input dim
#define G1    4096      // 4*D1
#define H2d   2048      // layer-2 hidden
#define G2    8192      // 4*H2d
#define K1    1024      // GEMM reduction dim (== D1)

typedef unsigned short ushort_t;
typedef unsigned int   uint32;

typedef __attribute__((ext_vector_type(8)))  __bf16 v8bf;
typedef __attribute__((ext_vector_type(16))) __bf16 v16bf;
typedef __attribute__((ext_vector_type(8)))  float  v8f;

// ---------------- helpers ----------------
__device__ __forceinline__ float bflo2f(uint32 p) {
    uint32 u = p << 16;
    return __builtin_bit_cast(float, u);
}
__device__ __forceinline__ float bfhi2f(uint32 p) {
    uint32 u = p & 0xFFFF0000u;
    return __builtin_bit_cast(float, u);
}
__device__ __forceinline__ ushort_t f2bf(float f) {
    uint32 u = __builtin_bit_cast(uint32, f);
    uint32 r = (u + 0x7FFFu + ((u >> 16) & 1u)) >> 16;   // RNE
    return (ushort_t)r;
}
__device__ __forceinline__ float sigmoidf(float x) {
    return 1.0f / (1.0f + __expf(-x));
}

// Software grid barrier for small persistent grids (all blocks co-resident).
__device__ __forceinline__ void grid_barrier(int* cnt, int* gen, int nwg) {
    __syncthreads();
    if (threadIdx.x == 0) {
        __threadfence();
        volatile int* vgen = gen;
        int g = *vgen;
        if (atomicAdd(cnt, 1) == nwg - 1) {
            *cnt = 0;
            __threadfence();
            atomicAdd(gen, 1);
        } else {
            while (*vgen == g) { __builtin_amdgcn_s_sleep(2); }
        }
        __threadfence();
    }
    __syncthreads();
}

// Vectorized bf16-weight x f32-LDS dot: nPairs4 = K/8 uint4 iterations.
__device__ __forceinline__ float dot_bf16_lds(const uint4* __restrict__ w4,
                                              const float* __restrict__ h_lds,
                                              int nPairs4) {
    float acc = 0.f;
#pragma unroll 4
    for (int k = 0; k < nPairs4; ++k) {
        uint4 p = w4[k];
        const float4* h4 = (const float4*)&h_lds[8 * k];
        float4 ha = h4[0];
        float4 hb = h4[1];
        acc += bflo2f(p.x) * ha.x + bfhi2f(p.x) * ha.y;
        acc += bflo2f(p.y) * ha.z + bfhi2f(p.y) * ha.w;
        acc += bflo2f(p.z) * hb.x + bfhi2f(p.z) * hb.y;
        acc += bflo2f(p.w) * hb.z + bfhi2f(p.w) * hb.w;
    }
    return acc;
}

// ---------------- kernels ----------------

__global__ void init_bar_kernel(int* bar) {
    if (threadIdx.x < 8) bar[threadIdx.x] = 0;
}

__global__ void cvt_f32_bf16_kernel(const float* __restrict__ in,
                                    ushort_t* __restrict__ out, int n) {
    for (int i = blockIdx.x * blockDim.x + threadIdx.x; i < n;
         i += gridDim.x * blockDim.x)
        out[i] = f2bf(in[i]);
}

// pre1[row] = b_ih1[row] + b_hh1[row] + dot(W_ih1[row,:], x)   (x constant over t)
__global__ void pre1_kernel(const float* __restrict__ x,
                            const float* __restrict__ Wih1,
                            const float* __restrict__ bih1,
                            const float* __restrict__ bhh1,
                            float* __restrict__ pre1) {
    __shared__ __align__(16) float xs[D1];
    for (int i = threadIdx.x; i < D1; i += blockDim.x) xs[i] = x[i];
    __syncthreads();
    int row = blockIdx.x * blockDim.x + threadIdx.x;   // 0..4095
    const float* w = Wih1 + (size_t)row * D1;
    float acc = bih1[row] + bhh1[row];
#pragma unroll 4
    for (int k = 0; k < D1; ++k) acc += w[k] * xs[k];
    pre1[row] = acc;
}

// Layer-1 recurrence: 16 blocks x 256 = 4096 threads, one gate-row each.
// Single grid barrier per step: gate combine is done redundantly per WG
// (c state lives in LDS), and zbuf is double-buffered by step parity.
__global__ void lstm1_kernel(const ushort_t* __restrict__ Whh1bf,
                             const float* __restrict__ pre1,
                             float* zbuf,                 // 2 x G1 (ping-pong)
                             ushort_t* __restrict__ h1bf, // T x D1 output (bf16)
                             int* cnt, int* gen) {
    __shared__ __align__(16) float h_lds[D1];
    __shared__ __align__(16) float c_lds[D1];
    const int t = blockIdx.x * blockDim.x + threadIdx.x;   // 0..4095
    for (int i = threadIdx.x; i < D1; i += blockDim.x) {
        h_lds[i] = 0.f;
        c_lds[i] = 0.f;
    }
    __syncthreads();

    const uint4* w4 = (const uint4*)(Whh1bf + (size_t)t * D1);
    const float p1 = pre1[t];

    for (int s = 0; s < T_SEQ; ++s) {
        float* zb = zbuf + (s & 1) * G1;
        zb[t] = p1 + dot_bf16_lds(w4, h_lds, D1 / 8);
        grid_barrier(cnt, gen, 16);

        // Redundant per-WG gate combine over all cells (i,f,g,o order).
        for (int j = threadIdx.x; j < D1; j += blockDim.x) {
            float zi = zb[j];
            float zf = zb[j + D1];
            float zg = zb[j + 2 * D1];
            float zo = zb[j + 3 * D1];
            float c = sigmoidf(zf) * c_lds[j] + sigmoidf(zi) * tanhf(zg);
            c_lds[j] = c;
            float h = sigmoidf(zo) * tanhf(c);
            h_lds[j] = h;
            if (blockIdx.x == 0) h1bf[(size_t)s * D1 + j] = f2bf(h);
        }
        __syncthreads();
    }
}

// pre2 = h1 (T x K1, bf16) * W_ih2^T (K1 x G2, bf16 rows k-contiguous) + bias
// One wave per 16x16 tile, 8 waves/block -> 32(M) x 64(N) block tile.
__global__ void gemm_pre2_kernel(const ushort_t* __restrict__ Abf,   // T x K1
                                 const ushort_t* __restrict__ Bbf,   // G2 x K1
                                 const float* __restrict__ bih2,
                                 const float* __restrict__ bhh2,
                                 float* __restrict__ pre2) {
    const int lane = threadIdx.x & 31;
    const int wave = threadIdx.x >> 5;        // 0..7
    const int wm = wave & 1, wn = wave >> 1;  // 2 x 4 wave grid
    const int row0 = blockIdx.y * 32 + wm * 16;   // time (M)
    const int col0 = blockIdx.x * 64 + wn * 16;   // gate-row (N)
    const int r   = lane & 15;
    const int sel = lane >> 4;

    const ushort_t* arow = Abf + (size_t)(row0 + r) * K1;
    const ushort_t* brow = Bbf + (size_t)(col0 + r) * K1;

    v8f acc = {};
    for (int kk = 0; kk < K1; kk += 32) {
        union { v16bf v; v8bf h[2]; } a, b;
        // 16-bit A layout: lane holds K = kk+sel*8..+7 and kk+16+sel*8..+7
        a.h[0] = *(const v8bf*)(arow + kk + sel * 8);
        a.h[1] = *(const v8bf*)(arow + kk + 16 + sel * 8);
        b.h[0] = *(const v8bf*)(brow + kk + sel * 8);
        b.h[1] = *(const v8bf*)(brow + kk + 16 + sel * 8);
        __builtin_prefetch(arow + kk + 128, 0, 3);
        __builtin_prefetch(brow + kk + 128, 0, 3);
        acc = __builtin_amdgcn_wmma_f32_16x16x32_bf16(
            /*neg_a=*/false, a.v, /*neg_b=*/false, b.v,
            /*c_mod=*/(short)0, acc, /*reuse_a=*/false, /*reuse_b=*/false);
    }

    const int col = col0 + r;
    const float bias = bih2[col] + bhh2[col];
#pragma unroll
    for (int e = 0; e < 8; ++e) {
        int row = row0 + 8 * sel + e;   // C layout: VGPR e -> M = e + 8*sel
        pre2[(size_t)row * G2 + col] = acc[e] + bias;
    }
}

// Layer-2 recurrence: 32 blocks x 256 = 8192 threads, one gate-row each.
// Same single-barrier / redundant-combine / ping-pong scheme as layer 1.
__global__ void lstm2_kernel(const ushort_t* __restrict__ Whh2bf,
                             const float* __restrict__ pre2,
                             float* zbuf,               // 2 x G2 (ping-pong)
                             float* __restrict__ h2s,   // T x H2d output (f32)
                             int* cnt, int* gen) {
    __shared__ __align__(16) float h_lds[H2d];
    __shared__ __align__(16) float c_lds[H2d];
    const int t = blockIdx.x * blockDim.x + threadIdx.x;   // 0..8191
    for (int i = threadIdx.x; i < H2d; i += blockDim.x) {
        h_lds[i] = 0.f;
        c_lds[i] = 0.f;
    }
    __syncthreads();

    const uint4* w4 = (const uint4*)(Whh2bf + (size_t)t * H2d);

    for (int s = 0; s < T_SEQ; ++s) {
        float* zb = zbuf + (s & 1) * G2;
        zb[t] = pre2[(size_t)s * G2 + t] + dot_bf16_lds(w4, h_lds, H2d / 8);
        grid_barrier(cnt, gen, 32);

        for (int j = threadIdx.x; j < H2d; j += blockDim.x) {
            float zi = zb[j];
            float zf = zb[j + H2d];
            float zg = zb[j + 2 * H2d];
            float zo = zb[j + 3 * H2d];
            float c = sigmoidf(zf) * c_lds[j] + sigmoidf(zi) * tanhf(zg);
            c_lds[j] = c;
            float h = sigmoidf(zo) * tanhf(c);
            h_lds[j] = h;
            if (blockIdx.x == 0) h2s[(size_t)s * H2d + j] = h;
        }
        __syncthreads();
    }
}

// out[t] = dot(h2s[t,:], W_out) + b_out
__global__ void out_kernel(const float* __restrict__ h2s,
                           const float* __restrict__ Wout,
                           const float* __restrict__ bout,
                           float* __restrict__ out) {
    __shared__ float red[256];
    const int t = blockIdx.x;
    const float* hrow = h2s + (size_t)t * H2d;
    float acc = 0.f;
    for (int k = threadIdx.x; k < H2d; k += 256) acc += hrow[k] * Wout[k];
    red[threadIdx.x] = acc;
    __syncthreads();
    for (int off = 128; off > 0; off >>= 1) {
        if (threadIdx.x < off) red[threadIdx.x] += red[threadIdx.x + off];
        __syncthreads();
    }
    if (threadIdx.x == 0) out[t] = red[0] + bout[0];
}

// ---------------- host-side launch ----------------
static inline size_t align_up(size_t x, size_t a) { return (x + a - 1) & ~(a - 1); }

extern "C" void kernel_launch(void* const* d_in, const int* in_sizes, int n_in,
                              void* d_out, int out_size, void* d_ws, size_t ws_size,
                              hipStream_t stream) {
    (void)in_sizes; (void)n_in; (void)out_size; (void)ws_size;

    const float* x     = (const float*)d_in[0];   // (1024)
    const float* Wih1  = (const float*)d_in[1];   // (4096,1024)
    const float* Whh1  = (const float*)d_in[2];   // (4096,1024)
    const float* bih1  = (const float*)d_in[3];   // (4096)
    const float* bhh1  = (const float*)d_in[4];   // (4096)
    const float* Wih2  = (const float*)d_in[5];   // (8192,1024)
    const float* Whh2  = (const float*)d_in[6];   // (8192,2048)
    const float* bih2  = (const float*)d_in[7];   // (8192)
    const float* bhh2  = (const float*)d_in[8];   // (8192)
    const float* Wout  = (const float*)d_in[9];   // (1,2048)
    const float* bout  = (const float*)d_in[10];  // (1)
    float* out = (float*)d_out;                   // (4096,1) fp32

    // ---- workspace carve-up ----
    char* ws = (char*)d_ws;
    size_t off = 0;
    auto carve = [&](size_t bytes) -> char* {
        char* p = ws + off;
        off = align_up(off + bytes, 256);
        return p;
    };
    int*      bar     = (int*)     carve(8 * sizeof(int));          // barrier state
    float*    pre1    = (float*)   carve((size_t)G1 * 4);
    float*    zbuf1   = (float*)   carve((size_t)2 * G1 * 4);       // ping-pong
    float*    zbuf2   = (float*)   carve((size_t)2 * G2 * 4);       // ping-pong
    ushort_t* Whh1bf  = (ushort_t*)carve((size_t)G1 * D1 * 2);      // 8 MB
    ushort_t* Wih2bf  = (ushort_t*)carve((size_t)G2 * K1 * 2);      // 16 MB
    ushort_t* Whh2bf  = (ushort_t*)carve((size_t)G2 * H2d * 2);     // 32 MB
    ushort_t* h1bf    = (ushort_t*)carve((size_t)T_SEQ * D1 * 2);   // 8 MB
    float*    pre2    = (float*)   carve((size_t)T_SEQ * G2 * 4);   // 128 MB
    float*    h2s     = (float*)   carve((size_t)T_SEQ * H2d * 4);  // 32 MB

    // 0) zero barrier state (ws is poisoned by the harness)
    init_bar_kernel<<<1, 32, 0, stream>>>(bar);

    // 1) weight conversions f32 -> bf16 (halves recurrent L2 streaming)
    cvt_f32_bf16_kernel<<<2048, 256, 0, stream>>>(Whh1, Whh1bf, G1 * D1);
    cvt_f32_bf16_kernel<<<2048, 256, 0, stream>>>(Wih2, Wih2bf, G2 * K1);
    cvt_f32_bf16_kernel<<<2048, 256, 0, stream>>>(Whh2, Whh2bf, G2 * H2d);

    // 2) input contribution (constant over timesteps)
    pre1_kernel<<<16, 256, 0, stream>>>(x, Wih1, bih1, bhh1, pre1);

    // 3) layer-1 recurrence (persistent, 1 grid barrier per step)
    lstm1_kernel<<<16, 256, 0, stream>>>(Whh1bf, pre1, zbuf1, h1bf,
                                         bar + 0, bar + 1);

    // 4) pre2 GEMM via bf16 WMMA: (4096x1024)*(1024x8192) + bias
    gemm_pre2_kernel<<<dim3(G2 / 64, T_SEQ / 32), 256, 0, stream>>>(
        h1bf, Wih2bf, bih2, bhh2, pre2);

    // 5) layer-2 recurrence (persistent, 1 grid barrier per step)
    lstm2_kernel<<<32, 256, 0, stream>>>(Whh2bf, pre2, zbuf2, h2s,
                                         bar + 2, bar + 3);

    // 6) output head
    out_kernel<<<T_SEQ, 256, 0, stream>>>(h2s, Wout, bout, out);
}
